// QLSTMPipelineStage_10247791968830
// MI455X (gfx1250) — matrix operand order repaired and checked by reading, not compiled, not run on hardware
//
#include <hip/hip_runtime.h>

// ---------------------------------------------------------------------------
// QLSTM pipeline forward for MI455X (gfx1250, wave32, WMMA bf16).
//   B=2, S=2048, D=1024, DFF=2048, V=32000, L=4
// GEMMs: bf16 WMMA 16x16x32 (f32 accum), copy-free ping-pong K loop (unroll
// 64), branchless templated epilogue with fused bias + activation (+resid).
// Weights converted to bf16 (145MB, L2-resident) and pre-warmed into L2 via
// gfx1250 async-to-LDS streaming. LSTM recurrence: chunked 3-phase scan.
// ---------------------------------------------------------------------------

#define B_    2
#define S_    2048
#define D_    1024
#define DFF_  2048
#define V_    32000
#define L_    4
#define M_    (B_ * S_)        // 4096 token rows
#define CH_   32               // scan chunks per sequence
#define CLEN_ (S_ / CH_)       // 64 steps per chunk
#define EPS_  1e-6f

typedef __attribute__((ext_vector_type(16))) __bf16 bf16x16;
typedef __attribute__((ext_vector_type(8)))  __bf16 bf16x8;
typedef __attribute__((ext_vector_type(8)))  float  f32x8;

__device__ __forceinline__ unsigned short f2bf(float f) {
  union { float f; unsigned int u; } cv; cv.f = f;
  unsigned int u = cv.u;
  unsigned int r = (u + 0x7FFFu + ((u >> 16) & 1u)) >> 16;  // RNE
  return (unsigned short)r;
}
__device__ __forceinline__ float sigmoidf_(float x) { return 1.f / (1.f + __expf(-x)); }

enum { ACT_NONE = 0, ACT_SIG = 1, ACT_TANH = 2, ACT_SILU = 3 };

template <int ACT>
__device__ __forceinline__ float epi_act(float v) {
  if constexpr (ACT == ACT_SIG)  return sigmoidf_(v);
  else if constexpr (ACT == ACT_TANH) return tanhf(v);
  else if constexpr (ACT == ACT_SILU) return v * sigmoidf_(v);
  else return v;
}

// ---------------------------------------------------------------------------
// WMMA GEMM: C[M,N] = act(A[M,K](bf16) * W[N,K]^T(bf16) + bias[N]) (+ resid)
// Block: 256 thr = 8 waves; wave tile 32x64 (2x4 WMMA 16x16 tiles);
// block tile 128x128. Requires M,N % 128 == 0, K % 64 == 0, K >= 128,
// bias non-null.
// ---------------------------------------------------------------------------
template <int ACT, bool RESID>
__global__ __launch_bounds__(256) void wmma_gemm_bf16(
    const unsigned short* __restrict__ A,   // [M,K] bf16 bits, K contiguous
    const unsigned short* __restrict__ W,   // [N,K] bf16 bits, K contiguous
    const float* __restrict__ bias,         // [N]
    const float* __restrict__ resid,        // [M,N] if RESID
    float* __restrict__ C, int M, int N, int K)
{
  const int lane = threadIdx.x & 31;
  const int wave = threadIdx.x >> 5;
  const int bm = blockIdx.y * 128 + (wave & 3) * 32;   // 4 waves along M
  const int bn = blockIdx.x * 128 + (wave >> 2) * 64;  // 2 waves along N
  const int lr = lane & 15;   // row (A) / col (B) within 16-tile
  const int lh = lane >> 4;   // K-half selector

  f32x8 acc[2][4] = {};

  const unsigned short* Ab = A + (size_t)bm * K;
  const unsigned short* Wb = W + (size_t)bn * K;

  // A operand (16x32): lane = row, 16 contiguous K at k + lh*16 (matches B).
  auto loadA = [&](int k, bf16x16 af[2]) {
#pragma unroll
    for (int i = 0; i < 2; ++i) {
      const unsigned short* p = Ab + (size_t)(i * 16 + lr) * K + k + lh * 16;
      af[i] = *(const bf16x16*)p;
    }
  };
  // B operand (32x16): lane = column, 16 contiguous K at k + lh*16.
  auto loadB = [&](int k, bf16x16 bfr[4]) {
#pragma unroll
    for (int j = 0; j < 4; ++j)
      bfr[j] = *(const bf16x16*)(Wb + (size_t)(j * 16 + lr) * K + k + lh * 16);
  };
  auto mma8 = [&](const bf16x16 af[2], const bf16x16 bfr[4]) {
#pragma unroll
    for (int i = 0; i < 2; ++i)
#pragma unroll
      for (int j = 0; j < 4; ++j)
        acc[i][j] = __builtin_amdgcn_wmma_f32_16x16x32_bf16(
            false, af[i], false, bfr[j], (short)0, acc[i][j], false, false);
  };

  // Copy-free ping-pong over 64-wide K super-steps (K % 64 == 0).
  bf16x16 a0[2], b0[4];
  loadA(0, a0);
  loadB(0, b0);
  int k = 0;
  for (; k + 64 < K; k += 64) {
    bf16x16 a1[2], b1[4];
    loadA(k + 32, a1);
    loadB(k + 32, b1);
    __builtin_prefetch(Ab + (size_t)lr * K + k + 256, 0, 0);  // global_prefetch_b8
    __builtin_prefetch(Wb + (size_t)lr * K + k + 256, 0, 0);
    mma8(a0, b0);
    loadA(k + 64, a0);
    loadB(k + 64, b0);
    mma8(a1, b1);
  }
  {
    bf16x16 a1[2], b1[4];
    loadA(k + 32, a1);
    loadB(k + 32, b1);
    mma8(a0, b0);
    mma8(a1, b1);
  }

  // Epilogue. C/D layout: reg r, lanes 0-15 -> (M=r, N=lane);
  // lanes 16-31 -> (M=r+8, N=lane-16). Branchless (templated).
  const size_t Ns = (size_t)N;
#pragma unroll
  for (int i = 0; i < 2; ++i) {
    const int row0 = bm + i * 16 + lh * 8;
#pragma unroll
    for (int j = 0; j < 4; ++j) {
      const int col = bn + j * 16 + lr;
      const float bj = bias[col];
      float* cp = C + (size_t)row0 * Ns + col;
      const float* rp = resid + (size_t)row0 * Ns + col;  // unused unless RESID
#pragma unroll
      for (int r = 0; r < 8; ++r) {
        float v = epi_act<ACT>(acc[i][j][r] + bj);
        if constexpr (RESID) v += rp[(size_t)r * Ns];
        cp[(size_t)r * Ns] = v;
      }
    }
  }
}

// ---------------------------------------------------------------------------
// L2 warm: stream a buffer through the gfx1250 async-to-LDS path so the
// weight working set is GL2-resident before the first GEMM touches it.
// VDST = LDS byte address (per-thread slot, never read back), VADDR = global
// address; tracked by ASYNCcnt (CDNA5 ISA).
// ---------------------------------------------------------------------------
__global__ __launch_bounds__(256) void l2_warm_async(
    const unsigned short* __restrict__ w, size_t n16 /* # of 16B chunks */) {
  __shared__ char smem[256 * 16];
  const unsigned lds_off = (unsigned)(size_t)&smem[threadIdx.x * 16];
  size_t t = (size_t)blockIdx.x * blockDim.x + threadIdx.x;
  const size_t stride = (size_t)gridDim.x * blockDim.x;
  for (; t < n16; t += stride) {
    const void* g = (const char*)w + t * 16;
    asm volatile("global_load_async_to_lds_b128 %0, %1, off"
                 :: "v"(lds_off), "v"(g) : "memory");
  }
  asm volatile("s_wait_asynccnt 0x0" ::: "memory");
}

// ---------------------------------------------------------------------------
// Elementwise / helper kernels
// ---------------------------------------------------------------------------
__global__ void cvt_f32_bf16_kernel(const float* __restrict__ src,
                                    unsigned short* __restrict__ dst, size_t n) {
  size_t t = (size_t)blockIdx.x * blockDim.x + threadIdx.x;
  const size_t stride = (size_t)gridDim.x * blockDim.x;
  for (; t < n; t += stride) dst[t] = f2bf(src[t]);
}

__global__ void embed_kernel(const int* __restrict__ x, const float* __restrict__ emb,
                             float* __restrict__ h) {
  const int m = blockIdx.x;
  const float* e = emb + (size_t)x[m] * D_;
  float* hr = h + (size_t)m * D_;
  for (int i = threadIdx.x; i < D_; i += blockDim.x) hr[i] = e[i];
}

__global__ __launch_bounds__(256) void rmsnorm_to_bf16(
    const float* __restrict__ x, const float* __restrict__ w,
    unsigned short* __restrict__ out) {
  const int row = blockIdx.x;
  const float* xr = x + (size_t)row * D_;
  float ss = 0.f;
  for (int i = threadIdx.x; i < D_; i += 256) { float v = xr[i]; ss += v * v; }
#pragma unroll
  for (int o = 16; o > 0; o >>= 1) ss += __shfl_down(ss, o, 32);
  __shared__ float red[8];
  __shared__ float s_scale;
  if ((threadIdx.x & 31) == 0) red[threadIdx.x >> 5] = ss;
  __syncthreads();
  if (threadIdx.x == 0) {
    float t = 0.f;
    for (int i = 0; i < 8; ++i) t += red[i];
    s_scale = rsqrtf(t / (float)D_ + EPS_);
  }
  __syncthreads();
  const float sc = s_scale;
  unsigned short* orow = out + (size_t)row * D_;
  for (int i = threadIdx.x; i < D_; i += 256) orow[i] = f2bf(xr[i] * sc * w[i]);
}

// a *= b  (i = tanh(ipre) * sigmoid(igpre); both already activated in GEMM)
__global__ void mul_inplace_kernel(float* __restrict__ a, const float* __restrict__ b,
                                   size_t n) {
  size_t t = (size_t)blockIdx.x * blockDim.x + threadIdx.x;
  const size_t stride = (size_t)gridDim.x * blockDim.x;
  for (; t < n; t += stride) a[t] *= b[t];
}

// out = bf16(a * b)  (SwiGLU combine; silu already applied in GEMM epilogue)
__global__ void mul_to_bf16_kernel(const float* __restrict__ a, const float* __restrict__ b,
                                   unsigned short* __restrict__ out, size_t n) {
  size_t t = (size_t)blockIdx.x * blockDim.x + threadIdx.x;
  const size_t stride = (size_t)gridDim.x * blockDim.x;
  for (; t < n; t += stride) out[t] = f2bf(a[t] * b[t]);
}

// Phase 1: per-chunk (prod f, local scan end). f,i already activated.
__global__ void scan_chunk_reduce(const float* __restrict__ fb, const float* __restrict__ ib,
                                  float* __restrict__ Ac, float* __restrict__ Bc) {
  const int t = blockIdx.x * blockDim.x + threadIdx.x;
  if (t >= B_ * D_ * CH_) return;
  const int d = t % D_;
  const int c = (t / D_) % CH_;
  const int b = t / (D_ * CH_);
  size_t base = ((size_t)b * S_ + (size_t)c * CLEN_) * D_ + d;
  float a = 1.f, bb = 0.f;
  for (int s = 0; s < CLEN_; ++s) {
    const size_t off = base + (size_t)s * D_;
    const float f = fb[off], iv = ib[off];
    bb = f * bb + iv;
    a *= f;
  }
  const size_t o = ((size_t)b * CH_ + c) * D_ + d;
  Ac[o] = a; Bc[o] = bb;
}

// Phase 2: sequential carry across the 32 chunk summaries per (b,d)
__global__ void scan_carry(const float* __restrict__ Ac, const float* __restrict__ Bc,
                           float* __restrict__ Hin, float* __restrict__ Pin) {
  const int t = blockIdx.x * blockDim.x + threadIdx.x;
  if (t >= B_ * D_) return;
  const int d = t % D_, b = t / D_;
  float h = 0.f, p = 1.f;
  for (int c = 0; c < CH_; ++c) {
    const size_t o = ((size_t)b * CH_ + c) * D_ + d;
    Hin[o] = h; Pin[o] = p;
    h = Ac[o] * h + Bc[o];
    p *= Ac[o];
  }
}

// Phase 3: replay chunks with carry-in, fuse output gate + residual.
// ogact already = sigmoid(og GEMM + bias).
__global__ void scan_apply(const float* __restrict__ fb, const float* __restrict__ ib,
                           const float* __restrict__ ogact, const float* __restrict__ Hin,
                           const float* __restrict__ Pin, const float* __restrict__ hinit,
                           float* __restrict__ h) {
  const int t = blockIdx.x * blockDim.x + threadIdx.x;
  if (t >= B_ * D_ * CH_) return;
  const int d = t % D_;
  const int c = (t / D_) % CH_;
  const int b = t / (D_ * CH_);
  const size_t o = ((size_t)b * CH_ + c) * D_ + d;
  float hp = Hin[o], p = Pin[o];
  const float hi0 = hinit[d];
  size_t base = ((size_t)b * S_ + (size_t)c * CLEN_) * D_ + d;
  for (int s = 0; s < CLEN_; ++s) {
    const size_t off = base + (size_t)s * D_;
    const float f = fb[off], iv = ib[off];
    hp = f * hp + iv;
    p *= f;
    const float hs = hp + hi0 * p;
    const float y = tanhf(hs) * ogact[off];
    h[off] += y;   // h currently holds the residual
  }
}

// ---------------------------------------------------------------------------
// Host orchestration
// ---------------------------------------------------------------------------
extern "C" void kernel_launch(void* const* d_in, const int* in_sizes, int n_in,
                              void* d_out, int out_size, void* d_ws, size_t ws_size,
                              hipStream_t stream) {
  (void)in_sizes; (void)n_in; (void)out_size; (void)ws_size;
  const int*   x     = (const int*)  d_in[0];
  const float* emb   = (const float*)d_in[1];
  const float* Wf    = (const float*)d_in[2];  const float* bfb  = (const float*)d_in[3];
  const float* Wi    = (const float*)d_in[4];  const float* bib  = (const float*)d_in[5];
  const float* Wig   = (const float*)d_in[6];  const float* bigb = (const float*)d_in[7];
  const float* Wog   = (const float*)d_in[8];  const float* bogb = (const float*)d_in[9];
  const float* W1    = (const float*)d_in[10]; const float* b1b  = (const float*)d_in[11];
  const float* Wa    = (const float*)d_in[12]; const float* bab  = (const float*)d_in[13];
  const float* Wo    = (const float*)d_in[14]; const float* bob  = (const float*)d_in[15];
  const float* nq    = (const float*)d_in[16]; const float* nf   = (const float*)d_in[17];
  const float* hinit = (const float*)d_in[18];
  const float* normw = (const float*)d_in[19];
  const float* Wout  = (const float*)d_in[20]; const float* bout = (const float*)d_in[21];
  float* out = (float*)d_out;

  // ---- workspace carve-up (256B aligned) ----
  char* wsp = (char*)d_ws;
  size_t cur = 0;
  auto alloc = [&](size_t bytes) -> void* {
    void* p = wsp + cur;
    cur = (cur + bytes + 255) & ~(size_t)255;
    return p;
  };
  const size_t MD = (size_t)M_ * D_, MDFF = (size_t)M_ * DFF_;
  float*          h    = (float*)alloc(MD * 4);
  unsigned short* xn   = (unsigned short*)alloc(MD * 2);
  float*          gbuf = (float*)alloc(4 * MD * 4);       // 4 gate bufs == 2 FFN bufs
  float *g0 = gbuf, *g1g = gbuf + MD, *g2g = gbuf + 2 * MD, *g3g = gbuf + 3 * MD;
  float *ga = gbuf, *gb = gbuf + MDFF;
  unsigned short* ffb  = (unsigned short*)alloc(MDFF * 2);
  const size_t SC = (size_t)B_ * D_ * CH_;
  float* Ac  = (float*)alloc(SC * 4);
  float* Bc  = (float*)alloc(SC * 4);
  float* Hin = (float*)alloc(SC * 4);
  float* Pin = (float*)alloc(SC * 4);
  const size_t szDD = (size_t)L_ * D_ * D_;
  const size_t szFD = (size_t)L_ * DFF_ * D_;
  unsigned short* WfB   = (unsigned short*)alloc(szDD * 2);
  unsigned short* WiB   = (unsigned short*)alloc(szDD * 2);
  unsigned short* WigB  = (unsigned short*)alloc(szDD * 2);
  unsigned short* WogB  = (unsigned short*)alloc(szDD * 2);
  unsigned short* W1B   = (unsigned short*)alloc(szFD * 2);
  unsigned short* WaB   = (unsigned short*)alloc(szFD * 2);
  unsigned short* WoB   = (unsigned short*)alloc(szFD * 2);
  unsigned short* WoutB = (unsigned short*)alloc((size_t)V_ * D_ * 2);

  auto cvt = [&](const float* s, unsigned short* d, size_t n) {
    cvt_f32_bf16_kernel<<<4096, 256, 0, stream>>>(s, d, n);
  };
  auto gemm = [&](int act, const unsigned short* Aa, const unsigned short* Ww,
                  const float* bias2, const float* resid, float* Cc,
                  int Mv, int Nv, int Kv) {
    dim3 grid(Nv / 128, Mv / 128);
    if (resid)
      wmma_gemm_bf16<ACT_NONE, true><<<grid, 256, 0, stream>>>(Aa, Ww, bias2, resid, Cc, Mv, Nv, Kv);
    else if (act == ACT_SIG)
      wmma_gemm_bf16<ACT_SIG, false><<<grid, 256, 0, stream>>>(Aa, Ww, bias2, nullptr, Cc, Mv, Nv, Kv);
    else if (act == ACT_TANH)
      wmma_gemm_bf16<ACT_TANH, false><<<grid, 256, 0, stream>>>(Aa, Ww, bias2, nullptr, Cc, Mv, Nv, Kv);
    else if (act == ACT_SILU)
      wmma_gemm_bf16<ACT_SILU, false><<<grid, 256, 0, stream>>>(Aa, Ww, bias2, nullptr, Cc, Mv, Nv, Kv);
    else
      wmma_gemm_bf16<ACT_NONE, false><<<grid, 256, 0, stream>>>(Aa, Ww, bias2, nullptr, Cc, Mv, Nv, Kv);
  };

  // ---- one-time (per launch) weight down-conversion: 145MB bf16, L2-resident ----
  cvt(Wf, WfB, szDD);  cvt(Wi, WiB, szDD);  cvt(Wig, WigB, szDD); cvt(Wog, WogB, szDD);
  cvt(W1, W1B, szFD);  cvt(Wa, WaB, szFD);  cvt(Wo, WoB, szFD);
  cvt(Wout, WoutB, (size_t)V_ * D_);
  // Pre-warm the whole converted bf16 weight region (WfB..WoutB contiguous)
  // into GL2 via the async-to-LDS path.
  {
    const size_t total_bf16_bytes =
        (szDD * 4 + szFD * 3 + (size_t)V_ * D_) * 2;
    l2_warm_async<<<2048, 256, 0, stream>>>(WfB, total_bf16_bytes / 16);
  }

  // ---- forward ----
  embed_kernel<<<M_, 256, 0, stream>>>(x, emb, h);

  for (int l = 0; l < L_; ++l) {
    const size_t od  = (size_t)l * D_ * D_;
    const size_t odf = (size_t)l * DFF_ * D_;
    const size_t odo = (size_t)l * D_ * DFF_;

    // QLSTM sub-block (activations fused into GEMM epilogues)
    rmsnorm_to_bf16<<<M_, 256, 0, stream>>>(h, nq + (size_t)l * D_, xn);
    gemm(ACT_SIG,  xn, WfB  + od, bfb  + (size_t)l * D_, nullptr, g0,  M_, D_, D_);  // f
    gemm(ACT_TANH, xn, WiB  + od, bib  + (size_t)l * D_, nullptr, g1g, M_, D_, D_);  // tanh(i)
    gemm(ACT_SIG,  xn, WigB + od, bigb + (size_t)l * D_, nullptr, g2g, M_, D_, D_);  // sig(ig)
    gemm(ACT_SIG,  xn, WogB + od, bogb + (size_t)l * D_, nullptr, g3g, M_, D_, D_);  // sig(og)
    mul_inplace_kernel<<<4096, 256, 0, stream>>>(g1g, g2g, MD);                      // i
    scan_chunk_reduce<<<(B_ * D_ * CH_ + 255) / 256, 256, 0, stream>>>(g0, g1g, Ac, Bc);
    scan_carry<<<(B_ * D_ + 255) / 256, 256, 0, stream>>>(Ac, Bc, Hin, Pin);
    scan_apply<<<(B_ * D_ * CH_ + 255) / 256, 256, 0, stream>>>(
        g0, g1g, g3g, Hin, Pin, hinit + (size_t)l * D_, h);

    // SwiGLU FFN sub-block
    rmsnorm_to_bf16<<<M_, 256, 0, stream>>>(h, nf + (size_t)l * D_, xn);
    gemm(ACT_NONE, xn, W1B + odf, b1b + (size_t)l * DFF_, nullptr, ga, M_, DFF_, D_);
    gemm(ACT_SILU, xn, WaB + odf, bab + (size_t)l * DFF_, nullptr, gb, M_, DFF_, D_);
    mul_to_bf16_kernel<<<4096, 256, 0, stream>>>(ga, gb, ffb, MDFF);
    gemm(ACT_NONE, ffb, WoB + odo, bob + (size_t)l * D_, h, h, M_, D_, DFF_);  // fused residual
  }

  // final norm + LM head
  rmsnorm_to_bf16<<<M_, 256, 0, stream>>>(h, normw, xn);
  gemm(ACT_NONE, xn, WoutB, bout, nullptr, out, M_, V_, D_);
}
